// PaiNNInteraction_36601711296775
// MI455X (gfx1250) — compile-verified, matrix operand order checked by reading, not run on hardware
//
#include <hip/hip_runtime.h>
#include <stdint.h>

#define NN 50000
#define EE 500000
#define FF 128
#define NRBF 20

// LDS row stride (uint16 units); stride/2 = 145 dwords (odd) -> conflict-free
#define HS0 290

typedef __attribute__((ext_vector_type(16))) __bf16 v16bf;
typedef __attribute__((ext_vector_type(8)))  float  v8f;

union Frag {
    v16bf    v;
    uint32_t u[8];
    uint4    q[2];
};

__device__ __forceinline__ uint16_t f2bf(float x) {
    union { float f; uint32_t u; } a; a.f = x;
    uint32_t r = a.u + 0x7FFFu + ((a.u >> 16) & 1u);   // round-to-nearest-even
    return (uint16_t)(r >> 16);
}

__device__ __forceinline__ float silu_f(float x) {
    return x / (1.0f + __expf(-x));
}

// Pack W [Kact x Nout] (row-major f32) into per-fragment bf16 B-operand layout.
// Fragment (kt,nt): 32 lanes x 16 halves; lane -> N = nt*16 + (lane&15);
// half h, j=h>>1: K = kt*32 + ((j&4)?16:0) + (lane>>4)*8 + (j&3)*2 + (h&1).
__global__ void pack_w(const float* __restrict__ W, uint16_t* __restrict__ out,
                       int Kact, int Nout, int ktiles, int ntiles) {
    int total = ktiles * ntiles * 512;
    for (int t = blockIdx.x * blockDim.x + threadIdx.x; t < total;
         t += gridDim.x * blockDim.x) {
        int frag = t >> 9;
        int rem  = t & 511;
        int lane = rem >> 4;
        int h    = rem & 15;
        int kt   = frag / ntiles;
        int nt   = frag - kt * ntiles;
        int n    = nt * 16 + (lane & 15);
        int g    = lane >> 4;
        int j    = h >> 1;
        int kl   = ((j & 4) ? 16 : 0) + g * 8 + (j & 3) * 2 + (h & 1);
        int k    = kt * 32 + kl;
        float val = (k < Kact) ? W[(size_t)k * Nout + n] : 0.0f;
        out[t] = f2bf(val);
    }
}

// out = [s (N*F) | v (N*3*F)] initialized to the residual inputs.
__global__ void init_out(const float* __restrict__ s, const float* __restrict__ v,
                         float* __restrict__ out) {
    const size_t nf = (size_t)NN * FF;
    const size_t total = nf * 4;
    for (size_t i = blockIdx.x * (size_t)blockDim.x + threadIdx.x; i < total;
         i += (size_t)gridDim.x * blockDim.x) {
        out[i] = (i < nf) ? s[i] : v[i - nf];
    }
}

// Load one 16x32 bf16 A fragment from an LDS activation tile (row-major,
// stride HS0 uint16 per row). Same (lane,half)->K mapping as pack_w.
__device__ __forceinline__ void load_a(const uint32_t* __restrict__ h32,
                                       int m16, int g, int kt, Frag& a) {
    int base = m16 * HS0 + kt * 32 + g * 8;   // even by construction
#pragma unroll
    for (int j = 0; j < 8; ++j) {
        int kl = ((j & 4) ? 16 : 0) + (j & 3) * 2;
        a.u[j] = h32[(base + kl) >> 1];
    }
}

__device__ __forceinline__ void load_b(const uint4* __restrict__ wq,
                                       int frag, int lane, Frag& b) {
    const uint4* p = wq + (size_t)frag * 64 + lane * 2;
    b.q[0] = p[0];
    b.q[1] = p[1];
}

#define WMMA_BF16(A, B, C)                                                    \
    __builtin_amdgcn_wmma_f32_16x16x32_bf16(false, (A), false, (B), (short)0, \
                                            (C), false, false)

__launch_bounds__(32)
__global__ void painn_edge(const float* __restrict__ s,
                           const float* __restrict__ pos,
                           const int*   __restrict__ ei,
                           const uint16_t* __restrict__ w1p,
                           const uint16_t* __restrict__ w2p,
                           const uint16_t* __restrict__ wsp,
                           const uint16_t* __restrict__ wvp,
                           const float* __restrict__ b1,
                           const float* __restrict__ b2,
                           const float* __restrict__ bs,
                           const float* __restrict__ bv,
                           const float* __restrict__ centers,
                           const float* __restrict__ widths,
                           float* __restrict__ out_s,
                           float* __restrict__ out_v) {
    // Single activation buffer per M-tile: h0, then h1, then h2 overlay it.
    // (Each GEMM's A-fragments are hoisted to VGPRs before its nt loop, so
    // the LDS tile is dead when the next layer's outputs are written.)
    __shared__ __align__(16) uint16_t hA[2][16 * HS0];
    __shared__ float dirs[32 * 3];
    __shared__ int   srcs[32];
    __shared__ int   dsts[32];

    const int lane = threadIdx.x;
    const int e0   = blockIdx.x * 32;       // 32 edges = two 16-row M-tiles
    const int m16  = lane & 15;
    const int g    = lane >> 4;

    // ---- per-edge geometry + RBF (all 32 lanes, one edge each) ----
    {
        int e  = e0 + lane;
        int sn = ei[e];
        int dn = ei[EE + e];
        srcs[lane] = sn;
        dsts[lane] = dn;
        float rx = pos[dn * 3 + 0] - pos[sn * 3 + 0];
        float ry = pos[dn * 3 + 1] - pos[sn * 3 + 1];
        float rz = pos[dn * 3 + 2] - pos[sn * 3 + 2];
        float dist = sqrtf(rx * rx + ry * ry + rz * rz);
        float inv  = dist > 0.0f ? 1.0f / dist : 0.0f;
        dirs[lane * 3 + 0] = rx * inv;
        dirs[lane * 3 + 1] = ry * inv;
        dirs[lane * 3 + 2] = rz * inv;
        uint16_t* row = hA[g] + m16 * HS0;
#pragma unroll
        for (int j = 0; j < NRBF; ++j) {
            float t = (dist - centers[j]) / (widths[j] + 1e-8f);
            row[256 + j] = f2bf(__expf(-t * t));
        }
#pragma unroll
        for (int j = 256 + NRBF; j < 288; ++j) row[j] = 0;  // K padding
    }
    __syncthreads();

    // ---- gather s[src], s[dst] rows -> h0 (bf16) ----
    for (int idx = lane; idx < 32 * FF; idx += 32) {
        int m = idx >> 7;
        int f = idx & (FF - 1);
        uint16_t* row = hA[m >> 4] + (m & 15) * HS0;
        row[f]      = f2bf(s[(size_t)srcs[m] * FF + f]);
        row[FF + f] = f2bf(s[(size_t)dsts[m] * FF + f]);
    }
    __syncthreads();

    const uint32_t* h32t0 = (const uint32_t*)hA[0];
    const uint32_t* h32t1 = (const uint32_t*)hA[1];
    const uint4* w1q = (const uint4*)w1p;
    const uint4* w2q = (const uint4*)w2p;
    const uint4* wsq = (const uint4*)wsp;
    const uint4* wvq = (const uint4*)wvp;

    Frag a0[9], a1[9];

    // ============ GEMM1: h1 = silu(h0 @ W1 + b1), K=288, N=256 =============
#pragma unroll
    for (int kt = 0; kt < 9; ++kt) { load_a(h32t0, m16, g, kt, a0[kt]);
                                     load_a(h32t1, m16, g, kt, a1[kt]); }
    __syncthreads();   // A-frags in regs; LDS tile may now be overwritten

#pragma unroll 1
    for (int nt = 0; nt < 16; ++nt) {
        float bias = b1[nt * 16 + m16];
        v8f c0, c1;
#pragma unroll
        for (int i = 0; i < 8; ++i) { c0[i] = 0.0f; c1[i] = 0.0f; }
        // depth-2 software-pipelined B stream
        Frag f0, f1;
        load_b(w1q, 0 * 16 + nt, lane, f0);
        load_b(w1q, 1 * 16 + nt, lane, f1);
#pragma unroll
        for (int kt = 0; kt < 9; ++kt) {
            Frag fn;
            if (kt + 2 < 9) load_b(w1q, (kt + 2) * 16 + nt, lane, fn);
            c0 = WMMA_BF16(a0[kt].v, f0.v, c0);
            c1 = WMMA_BF16(a1[kt].v, f0.v, c1);
            f0 = f1;
            f1 = fn;
        }
#pragma unroll
        for (int r = 0; r < 8; ++r) {
            int off = (r + 8 * g) * HS0 + nt * 16 + m16;
            hA[0][off] = f2bf(silu_f(c0[r] + bias));
            hA[1][off] = f2bf(silu_f(c1[r] + bias));
        }
    }
    __syncthreads();

    // ============ GEMM2: h2 = silu(h1 @ W2 + b2), K=256, N=256 =============
#pragma unroll
    for (int kt = 0; kt < 8; ++kt) { load_a(h32t0, m16, g, kt, a0[kt]);
                                     load_a(h32t1, m16, g, kt, a1[kt]); }
    __syncthreads();

#pragma unroll 1
    for (int nt = 0; nt < 16; ++nt) {
        float bias = b2[nt * 16 + m16];
        v8f c0, c1;
#pragma unroll
        for (int i = 0; i < 8; ++i) { c0[i] = 0.0f; c1[i] = 0.0f; }
        Frag f0, f1;
        load_b(w2q, 0 * 16 + nt, lane, f0);
        load_b(w2q, 1 * 16 + nt, lane, f1);
#pragma unroll
        for (int kt = 0; kt < 8; ++kt) {
            Frag fn;
            if (kt + 2 < 8) load_b(w2q, (kt + 2) * 16 + nt, lane, fn);
            c0 = WMMA_BF16(a0[kt].v, f0.v, c0);
            c1 = WMMA_BF16(a1[kt].v, f0.v, c1);
            f0 = f1;
            f1 = fn;
        }
#pragma unroll
        for (int r = 0; r < 8; ++r) {
            int off = (r + 8 * g) * HS0 + nt * 16 + m16;
            hA[0][off] = f2bf(silu_f(c0[r] + bias));
            hA[1][off] = f2bf(silu_f(c1[r] + bias));
        }
    }
    __syncthreads();

    // ====== GEMM3/4: ds = h2 @ Ws + bs ; dvmag = h2 @ Wv + bv, N=128 =======
#pragma unroll
    for (int kt = 0; kt < 8; ++kt) { load_a(h32t0, m16, g, kt, a0[kt]);
                                     load_a(h32t1, m16, g, kt, a1[kt]); }

#pragma unroll 1
    for (int nt = 0; nt < 8; ++nt) {
        float biasS = bs[nt * 16 + m16];
        float biasV = bv[nt * 16 + m16];
        v8f s0, s1, v0, v1;
#pragma unroll
        for (int i = 0; i < 8; ++i) { s0[i] = 0.0f; s1[i] = 0.0f;
                                      v0[i] = 0.0f; v1[i] = 0.0f; }
        // depth-1 prefetch of the (bS,bV) pair (4 WMMAs per pair already)
        Frag fS0, fV0;
        load_b(wsq, 0 * 8 + nt, lane, fS0);
        load_b(wvq, 0 * 8 + nt, lane, fV0);
#pragma unroll
        for (int kt = 0; kt < 8; ++kt) {
            Frag fSn, fVn;
            if (kt + 1 < 8) {
                load_b(wsq, (kt + 1) * 8 + nt, lane, fSn);
                load_b(wvq, (kt + 1) * 8 + nt, lane, fVn);
            }
            s0 = WMMA_BF16(a0[kt].v, fS0.v, s0);
            s1 = WMMA_BF16(a1[kt].v, fS0.v, s1);
            v0 = WMMA_BF16(a0[kt].v, fV0.v, v0);
            v1 = WMMA_BF16(a1[kt].v, fV0.v, v1);
            fS0 = fSn;
            fV0 = fVn;
        }
        int n = nt * 16 + m16;
#pragma unroll
        for (int t = 0; t < 2; ++t) {
#pragma unroll
            for (int r = 0; r < 8; ++r) {
                int m = t * 16 + r + 8 * g;
                int d = dsts[m];
                float dsv = (t ? s1[r] : s0[r]) + biasS;
                atomicAdd(out_s + (size_t)d * FF + n, dsv);
                float vm = (t ? v1[r] : v0[r]) + biasV;
                float* vp = out_v + (size_t)d * 3 * FF + n;
                atomicAdd(vp,          dirs[m * 3 + 0] * vm);
                atomicAdd(vp + FF,     dirs[m * 3 + 1] * vm);
                atomicAdd(vp + 2 * FF, dirs[m * 3 + 2] * vm);
            }
        }
    }
}

extern "C" void kernel_launch(void* const* d_in, const int* in_sizes, int n_in,
                              void* d_out, int out_size, void* d_ws, size_t ws_size,
                              hipStream_t stream) {
    (void)in_sizes; (void)n_in; (void)out_size; (void)ws_size;
    const float* s       = (const float*)d_in[0];
    const float* v       = (const float*)d_in[1];
    const float* pos     = (const float*)d_in[2];
    const int*   ei      = (const int*)d_in[3];
    const float* W1      = (const float*)d_in[4];
    const float* b1      = (const float*)d_in[5];
    const float* W2      = (const float*)d_in[6];
    const float* b2      = (const float*)d_in[7];
    const float* Ws      = (const float*)d_in[8];
    const float* bs      = (const float*)d_in[9];
    const float* Wv      = (const float*)d_in[10];
    const float* bv      = (const float*)d_in[11];
    const float* centers = (const float*)d_in[12];
    const float* widths  = (const float*)d_in[13];

    // workspace: packed bf16 weight fragments (~400 KB total, L2-resident)
    uint16_t* w1p = (uint16_t*)d_ws;        // 9*16*512
    uint16_t* w2p = w1p + 9 * 16 * 512;     // 8*16*512
    uint16_t* wsp = w2p + 8 * 16 * 512;     // 8*8*512
    uint16_t* wvp = wsp + 8 * 8 * 512;      // 8*8*512

    pack_w<<<288, 256, 0, stream>>>(W1, w1p, 2 * FF + NRBF, 256, 9, 16);
    pack_w<<<256, 256, 0, stream>>>(W2, w2p, 256, 256, 8, 16);
    pack_w<<<128, 256, 0, stream>>>(Ws, wsp, 256, 128, 8, 8);
    pack_w<<<128, 256, 0, stream>>>(Wv, wvp, 256, 128, 8, 8);

    float* out_s = (float*)d_out;
    float* out_v = out_s + (size_t)NN * FF;
    init_out<<<4096, 256, 0, stream>>>(s, v, out_s);

    painn_edge<<<EE / 32, 32, 0, stream>>>(s, pos, ei, w1p, w2p, wsp, wvp,
                                           b1, b2, bs, bv, centers, widths,
                                           out_s, out_v);
}
